// DecoderWithAttention_8280696946879
// MI455X (gfx1250) — matrix operand order repaired and checked by reading, not compiled, not run on hardware
//
#include <hip/hip_runtime.h>
#include <hip/hip_bf16.h>

// ---------- types ----------
typedef __attribute__((ext_vector_type(8)))  float  f32x8;
typedef __attribute__((ext_vector_type(8)))  __bf16 bf16x8;
typedef __attribute__((ext_vector_type(16))) __bf16 bf16x16;

// ---------- constants ----------
#define BB   64      // batch
#define EE   512     // encoder dim
#define DD   512     // lstm / decoder dim
#define AA   512     // attention dim
#define TT   16      // timesteps (W of encoder)
#define PP   256     // H*W positions
#define GG   2048    // 4*D gates
#define VV   5000    // vocab
#define VPAD 5008    // vocab padded to 16

// ---------- WMMA helpers (layouts per CDNA5 ISA 7.12.2) ----------
// A tile: 16x32 bf16, row-major source, lda in elements.
__device__ __forceinline__ bf16x16 load_a_tile(const __bf16* base, int lda, int lane) {
    int row = lane & 15;
    int kb  = (lane >> 4) << 3;           // 0 or 8
    const bf16x8* p0 = reinterpret_cast<const bf16x8*>(base + (long)row * lda + kb);
    const bf16x8* p1 = reinterpret_cast<const bf16x8*>(base + (long)row * lda + kb + 16);
    bf16x8 lo = *p0, hi = *p1;
    return __builtin_shufflevector(lo, hi, 0,1,2,3,4,5,6,7,8,9,10,11,12,13,14,15);
}
// B tile: 32x16 bf16 from a row-major N x K matrix (each row is one output column).
__device__ __forceinline__ bf16x16 load_b_tile(const __bf16* base, int ldb, int lane) {
    int col = lane & 15;
    int kb  = (lane >> 4) << 3;           // 0 or 8
    const bf16x8* p0 = reinterpret_cast<const bf16x8*>(base + (long)col * ldb + kb);
    const bf16x8* p1 = reinterpret_cast<const bf16x8*>(base + (long)col * ldb + kb + 16);
    bf16x8 lo = *p0, hi = *p1;
    return __builtin_shufflevector(lo, hi, 0,1,2,3,4,5,6,7,8,9,10,11,12,13,14,15);
}
__device__ __forceinline__ f32x8 wmma_bf16(bf16x16 a, bf16x16 b, f32x8 c) {
    return __builtin_amdgcn_wmma_f32_16x16x32_bf16(false, a, false, b, (short)0, c, false, false);
}

__device__ __forceinline__ float sigmoidf(float x) { return 1.f / (1.f + __expf(-x)); }
__device__ __forceinline__ float wave_sum(float v) {
    #pragma unroll
    for (int m = 16; m > 0; m >>= 1) v += __shfl_xor(v, m, 32);
    return v;
}
__device__ __forceinline__ float wave_max(float v) {
    #pragma unroll
    for (int m = 16; m > 0; m >>= 1) v = fmaxf(v, __shfl_xor(v, m, 32));
    return v;
}

// ---------- generic batched bf16 WMMA GEMM: C = A(MxK) * B(NxK)^T + bias ----------
__global__ void gemm_wmma(const __bf16* __restrict__ A, const __bf16* __restrict__ Bm,
                          const float* __restrict__ bias,
                          float* __restrict__ Cf, __bf16* __restrict__ Cb,
                          int M, int N, int K, int lda, int ldb, int ldc,
                          long sA, long sB, long sC, int tilesN)
{
    A  += (long)blockIdx.y * sA;
    Bm += (long)blockIdx.y * sB;
    long cOff = (long)blockIdx.y * sC;

    int wave = threadIdx.x >> 5;
    int lane = threadIdx.x & 31;
    int tile = blockIdx.x * (blockDim.x >> 5) + wave;
    int tilesM = M >> 4;
    if (tile >= tilesM * tilesN) return;
    int tm = tile / tilesN, tn = tile % tilesN;

    const __bf16* Abase = A  + (long)(tm * 16) * lda;
    const __bf16* Bbase = Bm + (long)(tn * 16) * ldb;

    f32x8 acc = {};
    for (int k0 = 0; k0 < K; k0 += 32) {
        bf16x16 a = load_a_tile(Abase + k0, lda, lane);
        bf16x16 b = load_b_tile(Bbase + k0, ldb, lane);
        acc = wmma_bf16(a, b, acc);
    }

    int n  = tn * 16 + (lane & 15);
    int mb = tm * 16 + ((lane >> 4) << 3);
    if (n < N) {
        float bv = bias ? bias[n] : 0.f;
        #pragma unroll
        for (int v = 0; v < 8; ++v) {
            float val = acc[v] + bv;
            long idx = cOff + (long)(mb + v) * ldc + n;
            if (Cf) Cf[idx] = val;
            if (Cb) Cb[idx] = (__bf16)val;
        }
    }
}

// ---------- persistent recurrent LSTM scan: one workgroup per direction ----------
// xproj holds x@Wih^T for all t (natural order). h kept in LDS (bf16), c in registers.
__global__ __launch_bounds__(1024) void lstm_scan(
    const float* __restrict__ xproj, const __bf16* __restrict__ Whh,
    const float* __restrict__ bih, const float* __restrict__ bhh,
    float* __restrict__ outF, int ldF, int offF,
    __bf16* __restrict__ outB, int ldB, int offB,
    int reverse)
{
    __shared__ __bf16 hS[BB * DD];                 // 64 KB

    int tid  = threadIdx.x;
    int wave = tid >> 5, lane = tid & 31;
    int tmQ  = wave >> 3;                          // row-tile 0..3 (16 batch rows each)
    int cg   = wave & 7;                           // column group

    for (int i = tid; i < BB * DD; i += 1024) hS[i] = (__bf16)0.f;

    float creg[4][8];
    #pragma unroll
    for (int j = 0; j < 4; ++j)
        #pragma unroll
        for (int v = 0; v < 8; ++v) creg[j][v] = 0.f;
    __syncthreads();

    int mrow = tmQ * 16 + ((lane >> 4) << 3);      // C-layout row base
    int ncol = lane & 15;

    for (int s = 0; s < TT; ++s) {
        int t = reverse ? (TT - 1 - s) : s;
        const float* xp = xproj + (long)t * BB * GG;
        float hnew[4][8];

        #pragma unroll
        for (int j = 0; j < 4; ++j) {
            int cn = (cg + 8 * j) * 16;            // column base within D
            float gv[4][8];
            #pragma unroll
            for (int g4 = 0; g4 < 4; ++g4) {       // i, f, g, o blocks
                int n0 = g4 * DD + cn;
                const __bf16* Bb = Whh + (long)n0 * DD;
                f32x8 acc = {};
                for (int k0 = 0; k0 < DD; k0 += 32) {
                    bf16x16 a = load_a_tile(hS + (long)tmQ * 16 * DD + k0, DD, lane);
                    bf16x16 b = load_b_tile(Bb + k0, DD, lane);
                    acc = wmma_bf16(a, b, acc);
                }
                int n = n0 + ncol;
                float bsum = bih[n] + bhh[n];
                #pragma unroll
                for (int v = 0; v < 8; ++v)
                    gv[g4][v] = acc[v] + xp[(long)(mrow + v) * GG + n] + bsum;
            }
            #pragma unroll
            for (int v = 0; v < 8; ++v) {
                float iv = sigmoidf(gv[0][v]);
                float fv = sigmoidf(gv[1][v]);
                float g2 = tanhf(gv[2][v]);
                float ov = sigmoidf(gv[3][v]);
                float c  = fv * creg[j][v] + iv * g2;
                creg[j][v] = c;
                hnew[j][v] = ov * tanhf(c);
            }
        }

        __syncthreads();   // all waves finished reading old h
        #pragma unroll
        for (int j = 0; j < 4; ++j) {
            int d = (cg + 8 * j) * 16 + ncol;
            #pragma unroll
            for (int v = 0; v < 8; ++v) {
                int m = mrow + v;
                float h = hnew[j][v];
                hS[m * DD + d] = (__bf16)h;
                outB[(long)t * BB * ldB + (long)m * ldB + offB + d] = (__bf16)h;
                if (outF)
                    outF[(long)t * BB * ldF + (long)m * ldF + offF + d] = h;
            }
        }
        __syncthreads();   // new h visible for next step
    }
}

// ---------- data prep ----------
__global__ void cvt_f32_bf16(const float* __restrict__ s, __bf16* __restrict__ d, long n) {
    long i = (long)blockIdx.x * blockDim.x + threadIdx.x;
    if (i < n) d[i] = (__bf16)s[i];
}
__global__ void cvt_pad_rows(const float* __restrict__ s, __bf16* __restrict__ d,
                             int rows, int padRows, int cols) {
    long i = (long)blockIdx.x * blockDim.x + threadIdx.x;
    if (i >= (long)padRows * cols) return;
    int r = (int)(i / cols);
    d[i] = (r < rows) ? (__bf16)s[i] : (__bf16)0.f;
}
// encB[b,e,p] = bf16(enc[b,e,p]);  featsHol[b,p,e] = same (transposed)
__global__ void prep_enc(const float* __restrict__ enc, __bf16* __restrict__ encB,
                         __bf16* __restrict__ featsHol) {
    long i = (long)blockIdx.x * blockDim.x + threadIdx.x;
    if (i >= (long)BB * EE * PP) return;
    int p = (int)(i & (PP - 1));
    long r = i >> 8;
    int e = (int)(r & (EE - 1));
    int b = (int)(r >> 9);
    float v = enc[i];
    encB[i] = (__bf16)v;
    featsHol[((long)b * PP + p) * EE + e] = (__bf16)v;
}
// feats[t,b,e] = mean_h enc[b,e,h*16+t]
__global__ void feats_mean(const float* __restrict__ enc, __bf16* __restrict__ feats) {
    int i = blockIdx.x * blockDim.x + threadIdx.x;
    if (i >= TT * BB * EE) return;
    int e = i & (EE - 1);
    int r = i >> 9;
    int b = r & (BB - 1);
    int t = r >> 6;
    const float* base = enc + ((long)b * EE + e) * PP + t;
    float sum = 0.f;
    #pragma unroll
    for (int h = 0; h < 16; ++h) sum += base[h * 16];
    feats[i] = (__bf16)(sum * (1.f / 16.f));
}

// ---------- attention pointwise ----------
// attS[t,b,p] = sum_a relu(att1[b,p,a] + att2[t,b,a]) * Wfull[a] + bfull
__global__ void att_score(const __bf16* __restrict__ att1, const float* __restrict__ att2,
                          const float* __restrict__ Wfull, const float* __restrict__ bfull,
                          float* __restrict__ attS) {
    int tb = blockIdx.x;                       // t*64 + b
    int b  = tb & (BB - 1);
    int p  = threadIdx.x;                      // 0..255
    const __bf16* a1 = att1 + ((long)b * PP + p) * AA;
    const float*  a2 = att2 + (long)tb * AA;
    float s = 0.f;
    for (int a = 0; a < AA; ++a) {
        float v = (float)a1[a] + a2[a];
        s += fmaxf(v, 0.f) * Wfull[a];
    }
    attS[(long)tb * PP + p] = s + bfull[0];
}
// softmax over P=256, one wave per (t,b); output bf16 alpha
__global__ void softmax_rows(const float* __restrict__ attS, __bf16* __restrict__ alpha) {
    int w = (blockIdx.x * blockDim.x + threadIdx.x) >> 5;
    int lane = threadIdx.x & 31;
    if (w >= TT * BB) return;
    const float* row = attS + (long)w * PP;
    float vals[8];
    float mx = -1e30f;
    #pragma unroll
    for (int i = 0; i < 8; ++i) { vals[i] = row[lane + 32 * i]; mx = fmaxf(mx, vals[i]); }
    mx = wave_max(mx);
    float sum = 0.f;
    #pragma unroll
    for (int i = 0; i < 8; ++i) { vals[i] = __expf(vals[i] - mx); sum += vals[i]; }
    sum = wave_sum(sum);
    float inv = 1.f / sum;
    #pragma unroll
    for (int i = 0; i < 8; ++i)
        alpha[(long)w * PP + lane + 32 * i] = (__bf16)(vals[i] * inv);
}
// gate softmax + fc_in assembly; one wave per (t,b)
__global__ void gate_fcin(const float* __restrict__ hid, const float* __restrict__ awe,
                          const float* __restrict__ Wg, const float* __restrict__ bg,
                          __bf16* __restrict__ fcin) {
    int w = (blockIdx.x * blockDim.x + threadIdx.x) >> 5;
    int lane = threadIdx.x & 31;
    if (w >= TT * BB) return;
    const float* h  = hid + (long)w * 1024;
    const float* aw = awe + (long)w * EE;
    float d0 = 0.f, d1 = 0.f;
    for (int i = 0; i < 48; ++i) {
        int idx = lane + 32 * i;
        float v = (idx < 1024) ? h[idx] : aw[idx - 1024];
        d0 += v * Wg[idx];
        d1 += v * Wg[1536 + idx];
    }
    d0 = wave_sum(d0) + bg[0];
    d1 = wave_sum(d1) + bg[1];
    float m = fmaxf(d0, d1);
    float e0 = __expf(d0 - m), e1 = __expf(d1 - m);
    float g0 = e0 / (e0 + e1), g1 = e1 / (e0 + e1);
    __bf16* o = fcin + (long)w * 1536;
    for (int i = 0; i < 48; ++i) {
        int idx = lane + 32 * i;
        float v = (idx < 1024) ? g0 * h[idx] : g1 * aw[idx - 1024];
        o[idx] = (__bf16)v;
    }
}

// ---------- host ----------
extern "C" void kernel_launch(void* const* d_in, const int* in_sizes, int n_in,
                              void* d_out, int out_size, void* d_ws, size_t ws_size,
                              hipStream_t stream) {
    const float* enc   = (const float*)d_in[0];
    const float* Wih1  = (const float*)d_in[1];
    const float* Whh1  = (const float*)d_in[2];
    const float* bih1  = (const float*)d_in[3];
    const float* bhh1  = (const float*)d_in[4];
    const float* Wih1r = (const float*)d_in[5];
    const float* Whh1r = (const float*)d_in[6];
    const float* bih1r = (const float*)d_in[7];
    const float* bhh1r = (const float*)d_in[8];
    const float* Wih2  = (const float*)d_in[9];
    const float* Whh2  = (const float*)d_in[10];
    const float* bih2  = (const float*)d_in[11];
    const float* bhh2  = (const float*)d_in[12];
    const float* Wih2r = (const float*)d_in[13];
    const float* Whh2r = (const float*)d_in[14];
    const float* bih2r = (const float*)d_in[15];
    const float* bhh2r = (const float*)d_in[16];
    const float* Wenc  = (const float*)d_in[17];
    const float* benc  = (const float*)d_in[18];
    const float* Wdec  = (const float*)d_in[19];
    const float* bdec  = (const float*)d_in[20];
    const float* Wfull = (const float*)d_in[21];
    const float* bfull = (const float*)d_in[22];
    const float* Wg    = (const float*)d_in[23];
    const float* bg    = (const float*)d_in[24];
    const float* Wfc   = (const float*)d_in[25];
    const float* bfc   = (const float*)d_in[26];
    float* out = (float*)d_out;

    // workspace allocator (256-B aligned offsets)
    char* ws = (char*)d_ws;
    size_t off = 0;
    auto alloc = [&](size_t bytes) -> char* {
        char* p = ws + off;
        off += (bytes + 255) & ~(size_t)255;
        return p;
    };
    __bf16* encB    = (__bf16*)alloc((size_t)BB * EE * PP * 2);
    __bf16* fHolB   = (__bf16*)alloc((size_t)BB * PP * EE * 2);
    __bf16* featsB  = (__bf16*)alloc((size_t)TT * BB * EE * 2);
    __bf16* Wih1b   = (__bf16*)alloc((size_t)GG * EE * 2);
    __bf16* Whh1b   = (__bf16*)alloc((size_t)GG * DD * 2);
    __bf16* Wih1rb  = (__bf16*)alloc((size_t)GG * EE * 2);
    __bf16* Whh1rb  = (__bf16*)alloc((size_t)GG * DD * 2);
    __bf16* Wih2b   = (__bf16*)alloc((size_t)GG * 1024 * 2);
    __bf16* Wih2rb  = (__bf16*)alloc((size_t)GG * 1024 * 2);
    __bf16* Whh2b   = (__bf16*)alloc((size_t)GG * DD * 2);
    __bf16* Whh2rb  = (__bf16*)alloc((size_t)GG * DD * 2);
    __bf16* Wencb   = (__bf16*)alloc((size_t)AA * EE * 2);
    __bf16* Wdecb   = (__bf16*)alloc((size_t)AA * 1024 * 2);
    __bf16* Wfcb    = (__bf16*)alloc((size_t)VPAD * 1536 * 2);
    float*  xp1f    = (float*)alloc((size_t)TT * BB * GG * 4);
    float*  xp1r    = (float*)alloc((size_t)TT * BB * GG * 4);
    float*  xp2f    = (float*)alloc((size_t)TT * BB * GG * 4);
    float*  xp2r    = (float*)alloc((size_t)TT * BB * GG * 4);
    __bf16* x2b     = (__bf16*)alloc((size_t)TT * BB * 1024 * 2);
    float*  hidF    = (float*)alloc((size_t)TT * BB * 1024 * 4);
    __bf16* hidB    = (__bf16*)alloc((size_t)TT * BB * 1024 * 2);
    __bf16* att1b   = (__bf16*)alloc((size_t)BB * PP * AA * 2);
    float*  att2f   = (float*)alloc((size_t)TT * BB * AA * 4);
    float*  attS    = (float*)alloc((size_t)TT * BB * PP * 4);
    __bf16* alphab  = (__bf16*)alloc((size_t)TT * BB * PP * 2);
    float*  aweF    = (float*)alloc((size_t)TT * BB * EE * 4);
    __bf16* fcinB   = (__bf16*)alloc((size_t)TT * BB * 1536 * 2);
    (void)ws_size; (void)n_in; (void)in_sizes; (void)out_size;

    auto cvt = [&](const float* s, __bf16* d, long n) {
        cvt_f32_bf16<<<(unsigned)((n + 255) / 256), 256, 0, stream>>>(s, d, n);
    };
    auto gemm = [&](const __bf16* A, const __bf16* Bm, const float* bias,
                    float* Cf, __bf16* Cb, int M, int N, int K,
                    int lda, int ldb, int ldc, long sA, long sB, long sC, int batch) {
        int tilesN = (N + 15) / 16;
        int tiles  = (M / 16) * tilesN;
        dim3 grid((unsigned)((tiles + 3) / 4), (unsigned)batch);
        gemm_wmma<<<grid, 128, 0, stream>>>(A, Bm, bias, Cf, Cb, M, N, K,
                                            lda, ldb, ldc, sA, sB, sC, tilesN);
    };

    // ---- 1) weight conversion ----
    cvt(Wih1,  Wih1b,  (long)GG * EE);
    cvt(Whh1,  Whh1b,  (long)GG * DD);
    cvt(Wih1r, Wih1rb, (long)GG * EE);
    cvt(Whh1r, Whh1rb, (long)GG * DD);
    cvt(Wih2,  Wih2b,  (long)GG * 1024);
    cvt(Wih2r, Wih2rb, (long)GG * 1024);
    cvt(Whh2,  Whh2b,  (long)GG * DD);
    cvt(Whh2r, Whh2rb, (long)GG * DD);
    cvt(Wenc,  Wencb,  (long)AA * EE);
    cvt(Wdec,  Wdecb,  (long)AA * 1024);
    {
        long n = (long)VPAD * 1536;
        cvt_pad_rows<<<(unsigned)((n + 255) / 256), 256, 0, stream>>>(Wfc, Wfcb, VV, VPAD, 1536);
    }

    // ---- 2) encoder prep ----
    {
        long n = (long)BB * EE * PP;
        prep_enc<<<(unsigned)((n + 255) / 256), 256, 0, stream>>>(enc, encB, fHolB);
        feats_mean<<<(TT * BB * EE + 255) / 256, 256, 0, stream>>>(enc, featsB);
    }

    // ---- 3) layer-1 input projections (time-parallel) ----
    gemm(featsB, Wih1b,  nullptr, xp1f, nullptr, TT * BB, GG, EE, EE, EE, GG, 0, 0, 0, 1);
    gemm(featsB, Wih1rb, nullptr, xp1r, nullptr, TT * BB, GG, EE, EE, EE, GG, 0, 0, 0, 1);

    // ---- 4) layer-1 recurrent scans (persistent workgroups) ----
    lstm_scan<<<1, 1024, 0, stream>>>(xp1f, Whh1b,  bih1,  bhh1,  nullptr, 0, 0, x2b, 1024, 0,   0);
    lstm_scan<<<1, 1024, 0, stream>>>(xp1r, Whh1rb, bih1r, bhh1r, nullptr, 0, 0, x2b, 1024, 512, 1);

    // ---- 5) layer-2 input projections + scans ----
    gemm(x2b, Wih2b,  nullptr, xp2f, nullptr, TT * BB, GG, 1024, 1024, 1024, GG, 0, 0, 0, 1);
    gemm(x2b, Wih2rb, nullptr, xp2r, nullptr, TT * BB, GG, 1024, 1024, 1024, GG, 0, 0, 0, 1);
    lstm_scan<<<1, 1024, 0, stream>>>(xp2f, Whh2b,  bih2,  bhh2,  hidF, 1024, 0,   hidB, 1024, 0,   0);
    lstm_scan<<<1, 1024, 0, stream>>>(xp2r, Whh2rb, bih2r, bhh2r, hidF, 1024, 512, hidB, 1024, 512, 1);

    // ---- 6) attention ----
    gemm(fHolB, Wencb, benc, nullptr, att1b, BB * PP, AA, EE, EE, EE, AA, 0, 0, 0, 1);
    gemm(hidB,  Wdecb, bdec, att2f, nullptr, TT * BB, AA, 1024, 1024, 1024, AA, 0, 0, 0, 1);
    att_score<<<TT * BB, PP, 0, stream>>>(att1b, att2f, Wfull, bfull, attS);
    softmax_rows<<<(TT * BB * 32 + 127) / 128, 128, 0, stream>>>(attS, alphab);
    // awe[t,b,:] = alpha[t,b,:] @ featsHol[b]   (batched over b; encB is the NxK B-operand)
    gemm(alphab, encB, nullptr, aweF, nullptr, TT, EE, PP,
         /*lda*/ BB * PP, /*ldb*/ PP, /*ldc*/ BB * EE,
         /*sA*/ PP, /*sB*/ (long)EE * PP, /*sC*/ EE, BB);

    // ---- 7) gate + fc head ----
    gate_fcin<<<(TT * BB * 32 + 127) / 128, 128, 0, stream>>>(hidF, aweF, Wg, bg, fcinB);
    gemm(fcinB, Wfcb, bfc, out, nullptr, TT * BB, VV, 1536, 1536, 1536, VV, 0, 0, 0, 1);
}
